// VELORA_34488587387269
// MI455X (gfx1250) — matrix operand order repaired and checked by reading, not compiled, not run on hardware
//
#include <hip/hip_runtime.h>
#include <math.h>

// ---------------------------------------------------------------------------
// Types for CDNA5 WMMA (wave32): v_wmma_f32_16x16x32_bf16
// ---------------------------------------------------------------------------
typedef __bf16        v16bf __attribute__((ext_vector_type(16)));
typedef float         v8f   __attribute__((ext_vector_type(8)));
typedef unsigned int  v4u   __attribute__((ext_vector_type(4)));

union Frag {            // 32 bytes: one WMMA A/B operand (8 VGPRs / lane)
  v16bf f;
  v4u   q[2];
};

constexpr int Bsz = 8192, Dd = 1024, HRr = 512, HEe = 4096, HFf = 2048;

// ---------------------------------------------------------------------------
// fp32 -> bf16 convert (row-major copy)
// ---------------------------------------------------------------------------
__global__ void velora_cvt_bf16(const float* __restrict__ src,
                                __bf16* __restrict__ dst, size_t n) {
  size_t i = (size_t)blockIdx.x * blockDim.x + threadIdx.x;
  if (i < n) dst[i] = (__bf16)src[i];
}

// fp32 [K][N] -> bf16 transposed [N][K]
__global__ void velora_cvt_transpose(const float* __restrict__ W,
                                     __bf16* __restrict__ Wt, int K, int N) {
  size_t i = (size_t)blockIdx.x * blockDim.x + threadIdx.x;
  if (i >= (size_t)K * N) return;
  int k = (int)(i / N);
  int n = (int)(i % N);
  Wt[(size_t)n * K + k] = (__bf16)W[i];
}

// ---------------------------------------------------------------------------
// Tiled bf16 WMMA GEMM:  C[M][N] = epilogue( A[M][K] @ Bt[N][K]^T + bias )
// Block tile 128x128, BK=32, 256 threads = 8 waves, wave tile 32x64.
// Double-buffered LDS staged by GLOBAL_LOAD_ASYNC_TO_LDS_B128 (ASYNCcnt).
// ---------------------------------------------------------------------------
#define GBM 128
#define GBN 128
#define GBK 32
#define LDST 48   // LDS row stride in bf16 elements (96B: 16B aligned, bank-skewed)

__global__ __launch_bounds__(256)
void velora_gemm_bf16_wmma(const __bf16* __restrict__ A,
                           const __bf16* __restrict__ Bt,
                           const float* __restrict__ bias,
                           float* __restrict__ outF,
                           __bf16* __restrict__ outB,
                           const float* __restrict__ residual,
                           int M, int N, int K, int relu) {
  __shared__ __align__(16) __bf16 sA[2][GBM * LDST];
  __shared__ __align__(16) __bf16 sB[2][GBN * LDST];

  const int tid  = threadIdx.x;
  const int lane = tid & 31;
  const int wid  = tid >> 5;
  const int wm   = wid & 3;          // 4 wave-rows of 32
  const int wn   = wid >> 2;         // 2 wave-cols of 64
  const int blockM = blockIdx.y * GBM;
  const int blockN = blockIdx.x * GBN;

  const int hsel = lane >> 4;        // half-wave select (K-interleave per ISA layout)
  const int ml   = lane & 15;        // row/col within 16

  v8f acc[2][4] = {};                // 8 accumulators x 8 VGPRs

  // Issue one K-slab of async memory->LDS copies (no VGPR round trip).
  auto stage = [&](int k0, int buf) {
    #pragma unroll
    for (int i = 0; i < 2; ++i) {
      int c    = tid + i * 256;      // 0..511 chunks of 8 bf16
      int row  = c >> 2;             // 0..127
      int col8 = (c & 3) << 3;       // 0,8,16,24
      unsigned long long ga =
          (unsigned long long)(size_t)(A  + (size_t)(blockM + row) * K + k0 + col8);
      unsigned long long gb =
          (unsigned long long)(size_t)(Bt + (size_t)(blockN + row) * K + k0 + col8);
      unsigned la = (unsigned)(size_t)&sA[buf][row * LDST + col8];
      unsigned lb = (unsigned)(size_t)&sB[buf][row * LDST + col8];
      asm volatile("global_load_async_to_lds_b128 %0, %1, off"
                   :: "v"(la), "v"(ga) : "memory");
      asm volatile("global_load_async_to_lds_b128 %0, %1, off"
                   :: "v"(lb), "v"(gb) : "memory");
    }
  };

  const int nk = K / GBK;
  stage(0, 0);
  asm volatile("s_wait_asynccnt 0" ::: "memory");
  __syncthreads();

  for (int ki = 0; ki < nk; ++ki) {
    const int buf = ki & 1;
    if (ki + 1 < nk) stage((ki + 1) * GBK, buf ^ 1);   // overlap DMA with WMMA

    // --- fragments per ISA 7.12.2 layouts ---
    Frag fa[2], fb[4];
    #pragma unroll
    for (int mt = 0; mt < 2; ++mt) {
      // A 16x32: lanes 0-15 hold K 0..7 & 16..23, lanes 16-31 hold K 8..15 & 24..31
      int r = wm * 32 + mt * 16 + ml;
      fa[mt].q[0] = *reinterpret_cast<const v4u*>(&sA[buf][r * LDST + hsel * 8]);
      fa[mt].q[1] = *reinterpret_cast<const v4u*>(&sA[buf][r * LDST + hsel * 8 + 16]);
    }
    #pragma unroll
    for (int nt = 0; nt < 4; ++nt) {
      // B 32x16: lanes 0-15 hold K 0..7 & 16..23, lanes 16-31 hold K 8..15 & 24..31 (col = lane&15)
      int cc = wn * 64 + nt * 16 + ml;
      fb[nt].q[0] = *reinterpret_cast<const v4u*>(&sB[buf][cc * LDST + hsel * 8]);
      fb[nt].q[1] = *reinterpret_cast<const v4u*>(&sB[buf][cc * LDST + hsel * 8 + 16]);
    }

    #pragma unroll
    for (int mt = 0; mt < 2; ++mt)
      #pragma unroll
      for (int nt = 0; nt < 4; ++nt)
        acc[mt][nt] = __builtin_amdgcn_wmma_f32_16x16x32_bf16(
            false, fa[mt].f, false, fb[nt].f,
            (short)0, acc[mt][nt], false, false);

    // next buffer's copies must land before anyone reads it next iteration
    asm volatile("s_wait_asynccnt 0" ::: "memory");
    __syncthreads();
  }

  // --- epilogue: bias (+ReLU) (+residual), f32 and/or bf16 stores ---
  #pragma unroll
  for (int mt = 0; mt < 2; ++mt) {
    #pragma unroll
    for (int nt = 0; nt < 4; ++nt) {
      int gn  = blockN + wn * 64 + nt * 16 + ml;
      float bb = bias[gn];
      int gm0 = blockM + wm * 32 + mt * 16 + hsel * 8;   // C/D: lanes 16-31 -> rows +8
      #pragma unroll
      for (int r = 0; r < 8; ++r) {
        float v = acc[mt][nt][r] + bb;
        if (relu) v = fmaxf(v, 0.0f);
        size_t gi = (size_t)(gm0 + r) * N + gn;
        if (residual) v += residual[gi];
        if (outF) outF[gi] = v;
        if (outB) outB[gi] = (__bf16)v;
      }
    }
  }
}

// ---------------------------------------------------------------------------
// Router head: one wave per row. dom logits, softmax weight, argmaxes.
// ---------------------------------------------------------------------------
__global__ __launch_bounds__(256)
void velora_router_post(const float* __restrict__ H,
                        const float* __restrict__ Wdom, const float* __restrict__ bdom,
                        const float* __restrict__ Wmop, const float* __restrict__ Wlt,
                        int* __restrict__ primary, float* __restrict__ wsel,
                        int* __restrict__ mop, int* __restrict__ ltask) {
  int lane = threadIdx.x & 31;
  int row  = blockIdx.x * 8 + (threadIdx.x >> 5);
  const float* h = H + (size_t)row * HRr;

  float d0 = 0.f, d1 = 0.f;
  float m0 = 0.f, m1 = 0.f, m2 = 0.f, m3 = 0.f;
  float l0 = 0.f, l1 = 0.f, l2 = 0.f, l3 = 0.f;
  for (int j = lane; j < HRr; j += 32) {
    float hv = h[j];
    d0 += hv * Wdom[j * 2 + 0]; d1 += hv * Wdom[j * 2 + 1];
    m0 += hv * Wmop[j * 4 + 0]; m1 += hv * Wmop[j * 4 + 1];
    m2 += hv * Wmop[j * 4 + 2]; m3 += hv * Wmop[j * 4 + 3];
    l0 += hv * Wlt[j * 4 + 0];  l1 += hv * Wlt[j * 4 + 1];
    l2 += hv * Wlt[j * 4 + 2];  l3 += hv * Wlt[j * 4 + 3];
  }
  #pragma unroll
  for (int off = 16; off > 0; off >>= 1) {
    d0 += __shfl_xor(d0, off); d1 += __shfl_xor(d1, off);
    m0 += __shfl_xor(m0, off); m1 += __shfl_xor(m1, off);
    m2 += __shfl_xor(m2, off); m3 += __shfl_xor(m3, off);
    l0 += __shfl_xor(l0, off); l1 += __shfl_xor(l1, off);
    l2 += __shfl_xor(l2, off); l3 += __shfl_xor(l3, off);
  }
  if (lane == 0) {
    d0 += bdom[0]; d1 += bdom[1];
    int p = (d1 > d0) ? 1 : 0;                       // argmax (first max wins)
    float mx = fmaxf(d0, d1);
    float e0 = expf(d0 - mx), e1 = expf(d1 - mx);
    float w  = ((p == 0) ? e0 : e1) / (e0 + e1);
    float mb[4] = {m0, m1, m2, m3};
    float lb[4] = {l0, l1, l2, l3};
    int mi = 0, li = 0;
    #pragma unroll
    for (int o = 1; o < 4; ++o) { if (mb[o] > mb[mi]) mi = o; if (lb[o] > lb[li]) li = o; }
    primary[row] = p; wsel[row] = w; mop[row] = mi; ltask[row] = li;
  }
}

// ---------------------------------------------------------------------------
// Expert inputs: x + embedding[idx], convert to bf16.
// ---------------------------------------------------------------------------
__global__ void velora_build_inputs(const float* __restrict__ x,
                                    const float* __restrict__ op_emb,
                                    const float* __restrict__ task_emb,
                                    const int* __restrict__ mop,
                                    const int* __restrict__ ltask,
                                    __bf16* __restrict__ mathin,
                                    __bf16* __restrict__ langin) {
  size_t i = (size_t)blockIdx.x * blockDim.x + threadIdx.x;
  if (i >= (size_t)Bsz * Dd) return;
  int b = (int)(i >> 10);
  int d = (int)(i & (Dd - 1));
  float xv = x[i];
  mathin[i] = (__bf16)(xv + op_emb[(size_t)mop[b] * Dd + d]);
  langin[i] = (__bf16)(xv + task_emb[(size_t)ltask[b] * Dd + d]);
}

// ---------------------------------------------------------------------------
// Select + scale: fused = w * (primary==0 ? math_out : lang_out)
// ---------------------------------------------------------------------------
__global__ void velora_select(const float* __restrict__ math_out,
                              const float* __restrict__ lang_out,
                              const int* __restrict__ primary,
                              const float* __restrict__ wsel,
                              float* __restrict__ fusedF,
                              __bf16* __restrict__ fusedB) {
  size_t i = (size_t)blockIdx.x * blockDim.x + threadIdx.x;
  if (i >= (size_t)Bsz * Dd) return;
  int b = (int)(i >> 10);
  float v = (primary[b] == 0) ? math_out[i] : lang_out[i];
  float f = wsel[b] * v;
  fusedF[i] = f;
  fusedB[i] = (__bf16)f;
}

// ---------------------------------------------------------------------------
// Host orchestration
// ---------------------------------------------------------------------------
extern "C" void kernel_launch(void* const* d_in, const int* in_sizes, int n_in,
                              void* d_out, int out_size, void* d_ws, size_t ws_size,
                              hipStream_t stream) {
  (void)in_sizes; (void)n_in; (void)out_size; (void)ws_size;
  const float* x      = (const float*)d_in[0];
  const float* W_r1   = (const float*)d_in[1];
  const float* b_r1   = (const float*)d_in[2];
  const float* W_dom  = (const float*)d_in[3];
  const float* b_dom  = (const float*)d_in[4];
  const float* W_mop  = (const float*)d_in[5];
  const float* W_lt   = (const float*)d_in[6];
  const float* op_emb = (const float*)d_in[7];
  const float* W_m1   = (const float*)d_in[8];
  const float* b_m1   = (const float*)d_in[9];
  const float* W_m2   = (const float*)d_in[10];
  const float* b_m2   = (const float*)d_in[11];
  const float* task_emb=(const float*)d_in[12];
  const float* W_l1   = (const float*)d_in[13];
  const float* b_l1   = (const float*)d_in[14];
  const float* W_l2   = (const float*)d_in[15];
  const float* b_l2   = (const float*)d_in[16];
  const float* W_f1   = (const float*)d_in[17];
  const float* b_f1   = (const float*)d_in[18];
  const float* W_f2   = (const float*)d_in[19];
  const float* b_f2   = (const float*)d_in[20];
  float* out = (float*)d_out;

  char* ws = (char*)d_ws;
  size_t off = 0;
  auto take = [&](size_t bytes) -> char* {
    char* p = ws + off;
    off += (bytes + 255) & ~(size_t)255;
    return p;
  };

  __bf16* xb      = (__bf16*)take((size_t)Bsz * Dd * 2);
  __bf16* mathin  = (__bf16*)take((size_t)Bsz * Dd * 2);
  __bf16* langin  = (__bf16*)take((size_t)Bsz * Dd * 2);
  __bf16* Wr1t    = (__bf16*)take((size_t)Dd * HRr * 2);   // [HR][D]
  __bf16* Wm1t    = (__bf16*)take((size_t)Dd * HEe * 2);   // [HE][D]
  __bf16* Wm2t    = (__bf16*)take((size_t)HEe * Dd * 2);   // [D][HE]
  __bf16* Wl1t    = (__bf16*)take((size_t)Dd * HEe * 2);
  __bf16* Wl2t    = (__bf16*)take((size_t)HEe * Dd * 2);
  __bf16* Wf1t    = (__bf16*)take((size_t)Dd * HFf * 2);   // [HF][D]
  __bf16* Wf2t    = (__bf16*)take((size_t)HFf * Dd * 2);   // [D][HF]
  float*  Hbuf    = (float*)take((size_t)Bsz * HRr * 4);
  int*    primary = (int*)take((size_t)Bsz * 4);
  float*  wsel    = (float*)take((size_t)Bsz * 4);
  int*    mop     = (int*)take((size_t)Bsz * 4);
  int*    ltask   = (int*)take((size_t)Bsz * 4);
  __bf16* hE      = (__bf16*)take((size_t)Bsz * HEe * 2);
  float*  mout    = (float*)take((size_t)Bsz * Dd * 4);
  float*  lout    = (float*)take((size_t)Bsz * Dd * 4);
  float*  fusedF  = (float*)take((size_t)Bsz * Dd * 4);
  __bf16* fusedB  = (__bf16*)take((size_t)Bsz * Dd * 2);
  __bf16* hF      = (__bf16*)take((size_t)Bsz * HFf * 2);

  const int T = 256;
  auto blocks = [&](size_t n) { return (unsigned)((n + T - 1) / T); };

  // 1) convert x, transpose+convert all GEMM weights to bf16 [N][K]
  velora_cvt_bf16<<<blocks((size_t)Bsz * Dd), T, 0, stream>>>(x, xb, (size_t)Bsz * Dd);
  velora_cvt_transpose<<<blocks((size_t)Dd * HRr), T, 0, stream>>>(W_r1, Wr1t, Dd, HRr);
  velora_cvt_transpose<<<blocks((size_t)Dd * HEe), T, 0, stream>>>(W_m1, Wm1t, Dd, HEe);
  velora_cvt_transpose<<<blocks((size_t)HEe * Dd), T, 0, stream>>>(W_m2, Wm2t, HEe, Dd);
  velora_cvt_transpose<<<blocks((size_t)Dd * HEe), T, 0, stream>>>(W_l1, Wl1t, Dd, HEe);
  velora_cvt_transpose<<<blocks((size_t)HEe * Dd), T, 0, stream>>>(W_l2, Wl2t, HEe, Dd);
  velora_cvt_transpose<<<blocks((size_t)Dd * HFf), T, 0, stream>>>(W_f1, Wf1t, Dd, HFf);
  velora_cvt_transpose<<<blocks((size_t)HFf * Dd), T, 0, stream>>>(W_f2, Wf2t, HFf, Dd);

  auto gemm = [&](const __bf16* A, const __bf16* Bt, const float* bias,
                  float* oF, __bf16* oB, const float* res,
                  int M, int N, int K, int relu) {
    dim3 grid(N / GBN, M / GBM);
    velora_gemm_bf16_wmma<<<grid, T, 0, stream>>>(A, Bt, bias, oF, oB, res, M, N, K, relu);
  };

  // 2) router: H = relu(x @ W_r1 + b_r1)
  gemm(xb, Wr1t, b_r1, Hbuf, nullptr, nullptr, Bsz, HRr, Dd, 1);
  // 3) router heads: softmax/argmax
  velora_router_post<<<Bsz / 8, T, 0, stream>>>(Hbuf, W_dom, b_dom, W_mop, W_lt,
                                                primary, wsel, mop, ltask);
  // 4) expert inputs
  velora_build_inputs<<<blocks((size_t)Bsz * Dd), T, 0, stream>>>(
      x, op_emb, task_emb, mop, ltask, mathin, langin);
  // 5) math expert
  gemm(mathin, Wm1t, b_m1, nullptr, hE, nullptr, Bsz, HEe, Dd, 1);
  gemm(hE, Wm2t, b_m2, mout, nullptr, nullptr, Bsz, Dd, HEe, 0);
  // 6) language expert (reuse hE)
  gemm(langin, Wl1t, b_l1, nullptr, hE, nullptr, Bsz, HEe, Dd, 1);
  gemm(hE, Wl2t, b_l2, lout, nullptr, nullptr, Bsz, Dd, HEe, 0);
  // 7) select + scale by routing weight
  velora_select<<<blocks((size_t)Bsz * Dd), T, 0, stream>>>(
      mout, lout, primary, wsel, fusedF, fusedB);
  // 8) fusion MLP with residual: out = fused + relu(fused@W_f1+b_f1)@W_f2 + b_f2
  gemm(fusedB, Wf1t, b_f1, nullptr, hF, nullptr, Bsz, HFf, Dd, 1);
  gemm(hF, Wf2t, b_f2, out, nullptr, fusedF, Bsz, Dd, HFf, 0);
}
